// MSDeformAttnPixelDecoder_18287970746412
// MI455X (gfx1250) — matrix operand order repaired and checked by reading, not compiled, not run on hardware
//
#include <hip/hip_runtime.h>
#include <math.h>

typedef __bf16 bf16_t;
typedef __attribute__((ext_vector_type(16))) __bf16 v16bf;
typedef __attribute__((ext_vector_type(8)))  float  v8f;
typedef __attribute__((ext_vector_type(4)))  int    v4i;

constexpr int CDIM   = 256;
constexpr int NHEADS = 8;
constexpr int NB     = 2;
constexpr int STOT   = 5376;          // 256 + 1024 + 4096
constexpr int BSR    = NB * STOT;     // 10752 rows

// GEMM flags
constexpr int GF_RELU  = 1;
constexpr int GF_ACCUM = 2;
constexpr int GF_CONV  = 4;

#if __has_builtin(__builtin_amdgcn_global_load_async_to_lds_b128) && \
    __has_builtin(__builtin_amdgcn_s_wait_asynccnt)
#define USE_ASYNC_LDS 1
// builtin expects (v4i __device__ *src, v4i __shared__ *dst, int offset, int cpol)
#define GLOBAL_V4I(p) ((__attribute__((address_space(1))) v4i*)(p))
#define LDS_V4I(p)    ((__attribute__((address_space(3))) v4i*)(p))
#else
#define USE_ASYNC_LDS 0
#endif

union FragAB { v16bf v; bf16_t e[16]; uint4 q2[2]; };

// ---------------------------------------------------------------------------
// Weight packer: packs a KxN (row-major, K=in, N=out) f32 weight into the
// CDNA5 bf16 B-fragment layout: tiles of 32(K)x16(N), 512 elems per tile,
// element index = tile*512 + lane*16 + e, where k = kt*32 + (lane>>4)*8 + (e&7) + (e>>3)*16,
// n = nt*16 + (lane&15).   trans=1 reads an (O=N, I=K[,3,3]) conv weight.
// ---------------------------------------------------------------------------
__global__ void k_pack(const float* __restrict__ w, bf16_t* __restrict__ dst,
                       int K, int N, int trans, int stride, int tap) {
  size_t idx = (size_t)blockIdx.x * blockDim.x + threadIdx.x;
  size_t n = (size_t)K * N;
  if (idx >= n) return;
  int tile = (int)(idx >> 9);
  int r    = (int)(idx & 511);
  int lane = r >> 4;
  int e    = r & 15;
  int nt16 = N >> 4;
  int nt   = tile % nt16;
  int kt   = tile / nt16;
  int k    = kt * 32 + (lane >> 4) * 8 + (e & 7) + (e >> 3) * 16;
  int nn   = nt * 16 + (lane & 15);
  size_t src = trans ? ((size_t)((size_t)nn * K + k) * stride + tap)
                     : ((size_t)((size_t)k * N + nn) * stride + tap);
  dst[idx] = (bf16_t)w[src];
}

// ---------------------------------------------------------------------------
// bf16 WMMA GEMM:  D[M,N] = act( A[M,K](bf16) @ Wp(packed bf16) + bias
//                               [+ resid] [+ D_prev] )
// One wave computes a 16 x (16*NSUB) tile, NSUB accumulators, with a
// software-pipelined (double-buffered) fragment loop + global prefetch so the
// WMMA issue overlaps the next step's loads.
// CONV mode: A rows are gathered from a spatially shifted (dy,dx) source pixel
// of an HWC image (zero outside), enabling 3x3 conv as 9 accumulated GEMMs.
// ---------------------------------------------------------------------------
template <int NSUB>
__global__ void k_gemm_wmma(const bf16_t* __restrict__ A, int lda,
                            const bf16_t* __restrict__ Wp,
                            const float* __restrict__ bias,
                            const float* __restrict__ resid,
                            float* __restrict__ D,
                            int M, int N, int K, int flags,
                            int convH, int convW, int cdy, int cdx) {
  int wave = (int)((blockIdx.x * blockDim.x + threadIdx.x) >> 5);
  int lane = threadIdx.x & 31;
  int ntn  = N / (16 * NSUB);
  int total = (M >> 4) * ntn;
  if (wave >= total) return;               // wave-uniform exit: EXEC stays all-1
  int tm = wave / ntn;
  int tn = wave % ntn;
  int row0 = tm << 4;
  int g    = lane >> 4;
  int lm   = lane & 15;
  int row  = row0 + lm;

  int  srow  = row;
  bool valid = true;
  if (flags & GF_CONV) {
    int hw  = convH * convW;
    int b   = row / hw;
    int pix = row - b * hw;
    int y   = pix / convW;
    int x   = pix - y * convW;
    int ys = y + cdy, xs = x + cdx;
    valid = (ys >= 0) && (ys < convH) && (xs >= 0) && (xs < convW);
    srow  = b * hw + (valid ? (ys * convW + xs) : 0);
  }

  v8f acc[NSUB] = {};
  const bf16_t* arow = A + (size_t)srow * lda + g * 8;
  int nt16 = N >> 4;
  size_t bstep = (size_t)nt16 * 512;       // packed elems per K-step (all N tiles)
  const bf16_t* bbase = Wp + (size_t)(tn * NSUB) * 512 + lane * 16;
  int ksteps = K >> 5;

  FragAB a0, a1, b0[NSUB], b1[NSUB];
  a0.q2[0] = make_uint4(0u, 0u, 0u, 0u);
  a0.q2[1] = make_uint4(0u, 0u, 0u, 0u);
  a1 = a0;

  // preload K-step 0 into set 0
  if (valid) {
    a0.q2[0] = *(const uint4*)(arow);
    a0.q2[1] = *(const uint4*)(arow + 16);
  }
#pragma unroll
  for (int s = 0; s < NSUB; ++s) {
    const uint4* p = (const uint4*)(bbase + (size_t)s * 512);
    b0[s].q2[0] = p[0]; b0[s].q2[1] = p[1];
  }

  for (int kt = 0; kt < ksteps; kt += 2) {
    if (kt + 1 < ksteps) {                 // prefetch step kt+1 into set 1
      if (valid) {
        a1.q2[0] = *(const uint4*)(arow + (kt + 1) * 32);
        a1.q2[1] = *(const uint4*)(arow + (kt + 1) * 32 + 16);
      }
#pragma unroll
      for (int s = 0; s < NSUB; ++s) {
        const uint4* p = (const uint4*)(bbase + (size_t)(kt + 1) * bstep + (size_t)s * 512);
        b1[s].q2[0] = p[0]; b1[s].q2[1] = p[1];
      }
    }
    if (kt + 2 < ksteps)                   // gfx1250 global_prefetch for kt+2
      __builtin_prefetch(bbase + (size_t)(kt + 2) * bstep, 0, 1);
#pragma unroll
    for (int s = 0; s < NSUB; ++s)
      acc[s] = __builtin_amdgcn_wmma_f32_16x16x32_bf16(false, a0.v, false, b0[s].v,
                                                       (short)0, acc[s], false, false);
    if (kt + 1 < ksteps) {
      if (kt + 2 < ksteps) {               // prefetch step kt+2 into set 0
        if (valid) {
          a0.q2[0] = *(const uint4*)(arow + (kt + 2) * 32);
          a0.q2[1] = *(const uint4*)(arow + (kt + 2) * 32 + 16);
        }
#pragma unroll
        for (int s = 0; s < NSUB; ++s) {
          const uint4* p = (const uint4*)(bbase + (size_t)(kt + 2) * bstep + (size_t)s * 512);
          b0[s].q2[0] = p[0]; b0[s].q2[1] = p[1];
        }
      }
#pragma unroll
      for (int s = 0; s < NSUB; ++s)
        acc[s] = __builtin_amdgcn_wmma_f32_16x16x32_bf16(false, a1.v, false, b1[s].v,
                                                         (short)0, acc[s], false, false);
    }
  }

  int colbase = tn * 16 * NSUB;
#pragma unroll
  for (int s = 0; s < NSUB; ++s) {
    int n = colbase + s * 16 + lm;
    float bs = bias ? bias[n] : 0.f;
#pragma unroll
    for (int r = 0; r < 8; ++r) {
      int m = row0 + g * 8 + r;            // C/D layout: VGPR r -> M = r + g*8
      size_t i0 = (size_t)m * N + n;
      float v = acc[s][r] + bs;
      if (resid) v += resid[i0];
      if (flags & GF_ACCUM) v += D[i0];
      if (flags & GF_RELU) v = fmaxf(v, 0.f);
      D[i0] = v;
    }
  }
}

// ---------------------------------------------------------------------------
// NCHW f32 -> (B*hw, ic) bf16 activation matrix, 32x32 LDS tile transpose.
// Coalesced along hw on the read side, along ic on the write side.
// ---------------------------------------------------------------------------
__global__ void k_feat2a_t(const float* __restrict__ f, bf16_t* __restrict__ a,
                           int ic, int hw) {
  __shared__ float t[32][36];              // stride 36: 16B-aligned rows, no bank conflict
  int pt_count = hw >> 5;
  int it_count = ic >> 5;
  int per_b = it_count * pt_count;
  int b  = blockIdx.x / per_b;
  int r2 = blockIdx.x - b * per_b;
  int it = r2 / pt_count;
  int pt = r2 - it * pt_count;
  int tx = threadIdx.x & 31;
  int ty = threadIdx.x >> 5;               // 0..7
  int i0 = it << 5, pix0 = pt << 5;
  const float* sb = f + ((size_t)b * ic + i0) * hw + pix0;
#if USE_ASYNC_LDS
  {
    int py  = threadIdx.x >> 3;            // 0..31 tile row (channel)
    int seg = threadIdx.x & 7;             // 8 x 16B segments per row
    const float* gp = sb + (size_t)py * hw + seg * 4;
    __builtin_amdgcn_global_load_async_to_lds_b128(
        GLOBAL_V4I(gp), LDS_V4I(&t[py][seg * 4]), 0, 0);
    __builtin_amdgcn_s_wait_asynccnt(0);
  }
#else
#pragma unroll
  for (int r = 0; r < 4; ++r) {
    int iy = ty + r * 8;
    t[iy][tx] = sb[(size_t)iy * hw + tx];
  }
#endif
  __syncthreads();
  bf16_t* db = a + ((size_t)b * hw + pix0) * ic + i0;
#pragma unroll
  for (int r = 0; r < 4; ++r) {
    int py = ty + r * 8;
    db[(size_t)py * ic + tx] = (bf16_t)t[tx][py];
  }
}

__global__ void k_tobf(const float* __restrict__ a, bf16_t* __restrict__ o, size_t n) {
  size_t i = (size_t)blockIdx.x * blockDim.x + threadIdx.x;
  if (i < n) o[i] = (bf16_t)a[i];
}

__global__ void k_addbf(const float* __restrict__ a, const float* __restrict__ b,
                        bf16_t* __restrict__ o, size_t n) {
  size_t i = (size_t)blockIdx.x * blockDim.x + threadIdx.x;
  if (i < n) o[i] = (bf16_t)(a[i] + b[i]);
}

// ---------------------------------------------------------------------------
// GroupNorm over an HWC slice: 32 groups of 8 channels, per batch.
// x points at batch-0 slice start; batch b at x + b*bstride.
// ---------------------------------------------------------------------------
__global__ void k_groupnorm(float* __restrict__ x, size_t bstride, int npix,
                            const float* __restrict__ gg, const float* __restrict__ bb,
                            int relu) {
  __shared__ float s1[256], s2[256];
  int b = blockIdx.x >> 5;
  int grp = blockIdx.x & 31;
  float* p = x + (size_t)b * bstride;
  int n = npix * 8;
  float sum = 0.f, sq = 0.f;
  for (int i = threadIdx.x; i < n; i += 256) {
    int pix = i >> 3; int c = grp * 8 + (i & 7);
    float v = p[(size_t)pix * CDIM + c];
    sum += v; sq += v * v;
  }
  s1[threadIdx.x] = sum; s2[threadIdx.x] = sq; __syncthreads();
  for (int st = 128; st > 0; st >>= 1) {
    if (threadIdx.x < st) { s1[threadIdx.x] += s1[threadIdx.x + st];
                            s2[threadIdx.x] += s2[threadIdx.x + st]; }
    __syncthreads();
  }
  float mean = s1[0] / n;
  float var  = s2[0] / n - mean * mean;
  float inv  = rsqrtf(var + 1e-5f);
  for (int i = threadIdx.x; i < n; i += 256) {
    int pix = i >> 3; int c = grp * 8 + (i & 7);
    size_t idx = (size_t)pix * CDIM + c;
    float v = (p[idx] - mean) * inv * gg[c] + bb[c];
    if (relu) v = fmaxf(v, 0.f);
    p[idx] = v;
  }
}

// LayerNorm with residual: out = LN(a + resid) over 256 features, 1 block/row.
__global__ void k_ln(const float* __restrict__ a, const float* __restrict__ res,
                     const float* __restrict__ gg, const float* __restrict__ bb,
                     float* __restrict__ out) {
  __shared__ float s1[256], s2[256];
  size_t row = blockIdx.x;
  int t = threadIdx.x;
  float v = a[row * CDIM + t] + (res ? res[row * CDIM + t] : 0.f);
  s1[t] = v; s2[t] = v * v; __syncthreads();
  for (int st = 128; st > 0; st >>= 1) {
    if (t < st) { s1[t] += s1[t + st]; s2[t] += s2[t + st]; }
    __syncthreads();
  }
  float m = s1[0] / CDIM;
  float var = s2[0] / CDIM - m * m;
  out[row * CDIM + t] = (v - m) * rsqrtf(var + 1e-5f) * gg[t] + bb[t];
}

// Sine positional encoding + level embed, written for both batches.
__global__ void k_pos(float* __restrict__ qpos, const float* __restrict__ emb,
                      int lvl, int h, int w, int start) {
  int idx = blockIdx.x * blockDim.x + threadIdx.x;
  int hw = h * w;
  if (idx >= hw * CDIM) return;
  int pix = idx / CDIM; int c = idx - pix * CDIM;
  int y = pix / w; int x = pix - y * w;
  const float scale = 6.283185307179586f;
  float val;
  if (c < 128) {
    float v = ((float)(y + 1) / ((float)h + 1e-6f)) * scale;
    float t = powf(10000.f, (float)(2 * (c >> 1)) / 128.f);
    val = (c & 1) ? cosf(v / t) : sinf(v / t);
  } else {
    int j = c - 128;
    float v = ((float)(x + 1) / ((float)w + 1e-6f)) * scale;
    float t = powf(10000.f, (float)(2 * (j >> 1)) / 128.f);
    val = (j & 1) ? cosf(v / t) : sinf(v / t);
  }
  val += emb[lvl * CDIM + c];
  for (int b = 0; b < NB; ++b)
    qpos[((size_t)(b * STOT + start + pix)) * CDIM + c] = val;
}

__global__ void k_ref(float* __restrict__ ref, int h, int w, int start) {
  int idx = blockIdx.x * blockDim.x + threadIdx.x;
  if (idx >= h * w) return;
  int y = idx / w; int x = idx - y * w;
  ref[(size_t)(start + idx) * 2 + 0] = (x + 0.5f) / w;
  ref[(size_t)(start + idx) * 2 + 1] = (y + 0.5f) / h;
}

// Softmax over the 12 (level,point) logits per (row, head), in place.
__global__ void k_softmax12(float* __restrict__ aw) {
  int i = blockIdx.x * blockDim.x + threadIdx.x;
  if (i >= BSR * NHEADS) return;
  float* p = aw + (size_t)(i / NHEADS) * 96 + (i % NHEADS) * 12;
  float mx = -1e30f;
  for (int j = 0; j < 12; ++j) mx = fmaxf(mx, p[j]);
  float s = 0.f;
  for (int j = 0; j < 12; ++j) { float e = expf(p[j] - mx); p[j] = e; s += e; }
  float inv = 1.f / s;
  for (int j = 0; j < 12; ++j) p[j] *= inv;
}

// Deformable sampling: one wave per (b, s, head), lane = channel-within-head.
// All 4 bilinear corner gathers are 128B coalesced loads (D=32 == wave32).
__global__ void k_sample(const float* __restrict__ v, const float* __restrict__ off,
                         const float* __restrict__ aw, const float* __restrict__ ref,
                         float* __restrict__ out) {
  int wave = (int)((blockIdx.x * blockDim.x + threadIdx.x) >> 5);
  int lane = threadIdx.x & 31;
  if (wave >= BSR * NHEADS) return;
  int row  = wave >> 3;              // b*STOT + s
  int head = wave & 7;
  int b = row / STOT;
  int s = row - b * STOT;
  const int lh[3]  = {16, 32, 64};
  const int lst[3] = {0, 256, 1280};
  float gx = ref[(size_t)s * 2];
  float gy = ref[(size_t)s * 2 + 1];
  float acc = 0.f;
  for (int l = 0; l < 3; ++l) {
    int h = lh[l], w = lh[l];
    const float* vl = v + ((size_t)(b * STOT + lst[l])) * CDIM + head * 32 + lane;
    const float* op = off + (size_t)row * 192 + (size_t)(head * 12 + l * 4) * 2;
    const float* ap = aw + (size_t)row * 96 + head * 12 + l * 4;
    for (int p = 0; p < 4; ++p) {
      float ox = op[p * 2], oy = op[p * 2 + 1];
      float a = ap[p];
      float x = (gx + ox / (float)w) * w - 0.5f;
      float y = (gy + oy / (float)h) * h - 0.5f;
      float x0f = floorf(x), y0f = floorf(y);
      float tx = x - x0f, ty = y - y0f;
      int x0 = (int)x0f, y0 = (int)y0f;
      int x1 = x0 + 1, y1 = y0 + 1;
      bool vx0 = (x0 >= 0) && (x0 < w), vx1 = (x1 >= 0) && (x1 < w);
      bool vy0 = (y0 >= 0) && (y0 < h), vy1 = (y1 >= 0) && (y1 < h);
      float s00 = (vx0 && vy0) ? vl[(size_t)(y0 * w + x0) * CDIM] : 0.f;
      float s10 = (vx1 && vy0) ? vl[(size_t)(y0 * w + x1) * CDIM] : 0.f;
      float s01 = (vx0 && vy1) ? vl[(size_t)(y1 * w + x0) * CDIM] : 0.f;
      float s11 = (vx1 && vy1) ? vl[(size_t)(y1 * w + x1) * CDIM] : 0.f;
      acc += a * (s00 * (1.f - tx) * (1.f - ty) + s10 * tx * (1.f - ty)
                + s01 * (1.f - tx) * ty + s11 * tx * ty);
    }
  }
  out[(size_t)row * CDIM + head * 32 + lane] = acc;
}

// Bilinear 64x64 -> 128x128 upsample of level-2 queries (HWC), added into lat.
__global__ void k_upadd(const float* __restrict__ q, float* __restrict__ lat) {
  size_t idx = (size_t)blockIdx.x * blockDim.x + threadIdx.x;
  size_t n = (size_t)NB * 16384 * CDIM;
  if (idx >= n) return;
  int b = (int)(idx / ((size_t)16384 * CDIM));
  size_t r = idx % ((size_t)16384 * CDIM);
  int pix = (int)(r / CDIM);
  int c   = (int)(r - (size_t)pix * CDIM);
  int y = pix >> 7, x = pix & 127;
  float sx = (x + 0.5f) * 0.5f - 0.5f;
  float sy = (y + 0.5f) * 0.5f - 0.5f;
  float x0f = floorf(sx), y0f = floorf(sy);
  float tx = sx - x0f, ty = sy - y0f;
  int x0 = (int)x0f, y0 = (int)y0f;
  int xa = min(max(x0, 0), 63), xb = min(max(x0 + 1, 0), 63);
  int ya = min(max(y0, 0), 63), yb = min(max(y0 + 1, 0), 63);
  const float* src = q + ((size_t)(b * STOT + 1280)) * CDIM + c;
  float v00 = src[(size_t)(ya * 64 + xa) * CDIM];
  float v10 = src[(size_t)(ya * 64 + xb) * CDIM];
  float v01 = src[(size_t)(yb * 64 + xa) * CDIM];
  float v11 = src[(size_t)(yb * 64 + xb) * CDIM];
  float up = v00 * (1.f - tx) * (1.f - ty) + v10 * tx * (1.f - ty)
           + v01 * (1.f - tx) * ty + v11 * tx * ty;
  lat[idx] += up;
}

// HWC -> NCHW via 32x32 LDS tile (coalesced read along C, write along pix).
// src rows at base + b*rowBatchStride; optional async global->LDS first phase.
__global__ void k_out_t(const float* __restrict__ src, size_t rowBatchStride,
                        float* __restrict__ dst, int hw) {
  __shared__ float t[32][36];
  int pt_count = hw >> 5;
  int per_b = pt_count * 8;                // 8 channel tiles of 32 in CDIM=256
  int b  = blockIdx.x / per_b;
  int r2 = blockIdx.x - b * per_b;
  int pt = r2 >> 3;
  int ct = r2 & 7;
  int tx = threadIdx.x & 31;
  int ty = threadIdx.x >> 5;
  int pix0 = pt << 5, c0 = ct << 5;
  const float* sb = src + ((size_t)b * rowBatchStride + pix0) * CDIM + c0;
#if USE_ASYNC_LDS
  {
    int py  = threadIdx.x >> 3;
    int seg = threadIdx.x & 7;
    const float* gp = sb + (size_t)py * CDIM + seg * 4;
    __builtin_amdgcn_global_load_async_to_lds_b128(
        GLOBAL_V4I(gp), LDS_V4I(&t[py][seg * 4]), 0, 0);
    __builtin_amdgcn_s_wait_asynccnt(0);
  }
#else
#pragma unroll
  for (int r = 0; r < 4; ++r) {
    int py = ty + r * 8;
    t[py][tx] = sb[(size_t)py * CDIM + tx];
  }
#endif
  __syncthreads();
  float* db = dst + ((size_t)b * CDIM + c0) * hw + pix0;
#pragma unroll
  for (int r = 0; r < 4; ++r) {
    int cy = ty + r * 8;
    db[(size_t)cy * hw + tx] = t[tx][cy];
  }
}

// ---------------------------------------------------------------------------
extern "C" void kernel_launch(void* const* d_in, const int* in_sizes, int n_in,
                              void* d_out, int out_size, void* d_ws, size_t ws_size,
                              hipStream_t stream) {
  (void)in_sizes; (void)n_in; (void)out_size; (void)ws_size;

  const float* feat[4];
  for (int i = 0; i < 4; ++i) feat[i] = (const float*)d_in[i];
  const float *icw[3], *icb[3], *icgg[3], *icgb[3];
  for (int i = 0; i < 3; ++i) {
    int b = 4 + 4 * i;
    icw[i]  = (const float*)d_in[b];     icb[i]  = (const float*)d_in[b + 1];
    icgg[i] = (const float*)d_in[b + 2]; icgb[i] = (const float*)d_in[b + 3];
  }
  const float* lvl_emb = (const float*)d_in[16];
  struct Layer { const float *vw,*vb,*ow,*ob,*aww,*awbv,*pw,*pb,*f1w,*f1b,*f2w,*f2b,*l1g,*l1b,*l2g,*l2b; } L[6];
  for (int l = 0; l < 6; ++l) {
    int b = 17 + 16 * l;
    L[l].vw  = (const float*)d_in[b + 0];  L[l].vb  = (const float*)d_in[b + 1];
    L[l].ow  = (const float*)d_in[b + 2];  L[l].ob  = (const float*)d_in[b + 3];
    L[l].aww = (const float*)d_in[b + 4];  L[l].awbv= (const float*)d_in[b + 5];
    L[l].pw  = (const float*)d_in[b + 6];  L[l].pb  = (const float*)d_in[b + 7];
    L[l].f1w = (const float*)d_in[b + 8];  L[l].f1b = (const float*)d_in[b + 9];
    L[l].f2w = (const float*)d_in[b + 10]; L[l].f2b = (const float*)d_in[b + 11];
    L[l].l1g = (const float*)d_in[b + 12]; L[l].l1b = (const float*)d_in[b + 13];
    L[l].l2g = (const float*)d_in[b + 14]; L[l].l2b = (const float*)d_in[b + 15];
  }
  const float* latw  = (const float*)d_in[113];
  const float* latgg = (const float*)d_in[114];
  const float* latgb = (const float*)d_in[115];
  const float* outw  = (const float*)d_in[116];
  const float* outgg = (const float*)d_in[117];
  const float* outgb = (const float*)d_in[118];
  const float* maskw = (const float*)d_in[119];
  const float* maskb = (const float*)d_in[120];

  // ---- workspace bump allocator ----
  char* wsb = (char*)d_ws;
  size_t wo = 0;
  auto alloc = [&](size_t bytes) -> void* {
    void* p = wsb + wo;
    wo = (wo + bytes + 255) & ~(size_t)255;
    return p;
  };
  float* q    = (float*)alloc((size_t)BSR * CDIM * 4);
  float* qpos = (float*)alloc((size_t)BSR * CDIM * 4);
  float* vbuf = (float*)alloc((size_t)BSR * CDIM * 4);
  float* attn = (float*)alloc((size_t)BSR * CDIM * 4);
  float* tmp  = (float*)alloc((size_t)BSR * CDIM * 4);
  float* offb = (float*)alloc((size_t)BSR * 192 * 4);
  float* awb  = (float*)alloc((size_t)BSR * 96 * 4);
  float* hid  = (float*)alloc((size_t)BSR * 1024 * 4);   // reused as maskout later
  float* refb = (float*)alloc((size_t)STOT * 2 * 4);
  bf16_t* qbf    = (bf16_t*)alloc((size_t)BSR * CDIM * 2);
  bf16_t* qqbf   = (bf16_t*)alloc((size_t)BSR * CDIM * 2);
  bf16_t* attnbf = (bf16_t*)alloc((size_t)BSR * CDIM * 2);
  bf16_t* hidbf  = (bf16_t*)alloc((size_t)BSR * 1024 * 2);
  float*  lat    = (float*)alloc((size_t)NB * 16384 * CDIM * 4); // lateral, lat+up, conv out
  bf16_t* yinbf  = (bf16_t*)alloc((size_t)NB * 16384 * CDIM * 2);
  bf16_t* convbf = (bf16_t*)alloc((size_t)NB * 16384 * CDIM * 2);
  bf16_t* featA  = (bf16_t*)alloc((size_t)NB * 16384 * 96 * 2); // largest A matrix
  const int icch[3] = {768, 384, 192};
  bf16_t* pk_ic[3];
  for (int i = 0; i < 3; ++i) pk_ic[i] = (bf16_t*)alloc((size_t)icch[i] * CDIM * 2);
  bf16_t* pk_lat = (bf16_t*)alloc((size_t)96 * CDIM * 2);
  bf16_t* pk_tap[9];
  for (int t = 0; t < 9; ++t) pk_tap[t] = (bf16_t*)alloc((size_t)CDIM * CDIM * 2);
  bf16_t* pk_mask = (bf16_t*)alloc((size_t)CDIM * CDIM * 2);
  struct PkL { bf16_t *v, *o, *a, *p, *f1, *f2; } pk[6];
  for (int l = 0; l < 6; ++l) {
    pk[l].v  = (bf16_t*)alloc((size_t)256 * 256 * 2);
    pk[l].o  = (bf16_t*)alloc((size_t)256 * 192 * 2);
    pk[l].a  = (bf16_t*)alloc((size_t)256 * 96 * 2);
    pk[l].p  = (bf16_t*)alloc((size_t)256 * 256 * 2);
    pk[l].f1 = (bf16_t*)alloc((size_t)256 * 1024 * 2);
    pk[l].f2 = (bf16_t*)alloc((size_t)1024 * 256 * 2);
  }
  float* maskout = hid;   // 44MB region >= 33.5MB needed, dead after encoder

  auto nblk = [](size_t n, int bs) { return (unsigned)((n + (size_t)bs - 1) / bs); };
  auto packw = [&](const float* src, bf16_t* dst, int K, int N, int trans, int stride, int tap) {
    size_t n = (size_t)K * N;
    k_pack<<<nblk(n, 256), 256, 0, stream>>>(src, dst, K, N, trans, stride, tap);
  };
  auto gemm = [&](const bf16_t* A, int lda, const bf16_t* Wp, const float* bias,
                  const float* resid, float* D, int M, int N, int K, int flags,
                  int cH, int cW, int dy, int dx) {
    if ((N & 63) == 0) {
      size_t waves = (size_t)(M / 16) * (N / 64);
      k_gemm_wmma<4><<<nblk(waves * 32, 128), 128, 0, stream>>>(A, lda, Wp, bias, resid, D,
                                                                M, N, K, flags, cH, cW, dy, dx);
    } else {
      size_t waves = (size_t)(M / 16) * (N / 32);
      k_gemm_wmma<2><<<nblk(waves * 32, 128), 128, 0, stream>>>(A, lda, Wp, bias, resid, D,
                                                                M, N, K, flags, cH, cW, dy, dx);
    }
  };
  auto feat2a = [&](const float* f, bf16_t* a, int ic, int hw) {
    k_feat2a_t<<<(unsigned)(NB * (ic / 32) * (hw / 32)), 256, 0, stream>>>(f, a, ic, hw);
  };
  auto outwrite = [&](const float* src, size_t rbs, float* dst, int hw) {
    k_out_t<<<(unsigned)(NB * (hw / 32) * 8), 256, 0, stream>>>(src, rbs, dst, hw);
  };

  // ---- pack all GEMM weights to bf16 fragment layout ----
  for (int i = 0; i < 3; ++i) packw(icw[i], pk_ic[i], icch[i], CDIM, 1, 1, 0);
  packw(latw, pk_lat, 96, CDIM, 1, 1, 0);
  for (int t = 0; t < 9; ++t) packw(outw, pk_tap[t], CDIM, CDIM, 1, 9, t);
  packw(maskw, pk_mask, CDIM, CDIM, 1, 1, 0);
  for (int l = 0; l < 6; ++l) {
    packw(L[l].vw,  pk[l].v,  256, 256, 0, 1, 0);
    packw(L[l].ow,  pk[l].o,  256, 192, 0, 1, 0);
    packw(L[l].aww, pk[l].a,  256, 96,  0, 1, 0);
    packw(L[l].pw,  pk[l].p,  256, 256, 0, 1, 0);
    packw(L[l].f1w, pk[l].f1, 256, 1024, 0, 1, 0);
    packw(L[l].f2w, pk[l].f2, 1024, 256, 0, 1, 0);
  }

  // ---- input projections + GN + positional encodings + reference points ----
  const int lh[3] = {16, 32, 64};
  const int lst[3] = {0, 256, 1280};
  for (int i = 0; i < 3; ++i) {
    int h = lh[i], w = lh[i], hw = h * w, ic = icch[i];
    feat2a(feat[3 - i], featA, ic, hw);
    for (int b = 0; b < NB; ++b)
      gemm(featA + (size_t)b * hw * ic, ic, pk_ic[i], icb[i], nullptr,
           q + ((size_t)(b * STOT + lst[i])) * CDIM, hw, CDIM, ic, 0, 0, 0, 0, 0);
    k_groupnorm<<<NB * 32, 256, 0, stream>>>(q + (size_t)lst[i] * CDIM,
                                             (size_t)STOT * CDIM, hw, icgg[i], icgb[i], 0);
    k_pos<<<nblk((size_t)hw * CDIM, 256), 256, 0, stream>>>(qpos, lvl_emb, i, h, w, lst[i]);
    k_ref<<<nblk((size_t)hw, 256), 256, 0, stream>>>(refb, h, w, lst[i]);
  }

  // ---- 6 encoder layers ----
  size_t nQC = (size_t)BSR * CDIM;
  for (int l = 0; l < 6; ++l) {
    k_tobf<<<nblk(nQC, 256), 256, 0, stream>>>(q, qbf, nQC);
    k_addbf<<<nblk(nQC, 256), 256, 0, stream>>>(q, qpos, qqbf, nQC);
    gemm(qbf, 256, pk[l].v, L[l].vb, nullptr, vbuf, BSR, 256, 256, 0, 0, 0, 0, 0);
    gemm(qqbf, 256, pk[l].o, L[l].ob, nullptr, offb, BSR, 192, 256, 0, 0, 0, 0, 0);
    gemm(qqbf, 256, pk[l].a, L[l].awbv, nullptr, awb, BSR, 96, 256, 0, 0, 0, 0, 0);
    k_softmax12<<<nblk((size_t)BSR * NHEADS, 256), 256, 0, stream>>>(awb);
    k_sample<<<nblk((size_t)BSR * NHEADS * 32, 256), 256, 0, stream>>>(vbuf, offb, awb, refb, attn);
    k_tobf<<<nblk(nQC, 256), 256, 0, stream>>>(attn, attnbf, nQC);
    gemm(attnbf, 256, pk[l].p, L[l].pb, nullptr, tmp, BSR, 256, 256, 0, 0, 0, 0, 0);
    k_ln<<<BSR, 256, 0, stream>>>(tmp, q, L[l].l1g, L[l].l1b, q);
    k_tobf<<<nblk(nQC, 256), 256, 0, stream>>>(q, qbf, nQC);
    gemm(qbf, 256, pk[l].f1, L[l].f1b, nullptr, hid, BSR, 1024, 256, GF_RELU, 0, 0, 0, 0);
    k_tobf<<<nblk((size_t)BSR * 1024, 256), 256, 0, stream>>>(hid, hidbf, (size_t)BSR * 1024);
    gemm(hidbf, 1024, pk[l].f2, L[l].f2b, nullptr, tmp, BSR, 256, 1024, 0, 0, 0, 0, 0);
    k_ln<<<BSR, 256, 0, stream>>>(tmp, q, L[l].l2g, L[l].l2b, q);
  }

  // ---- outputs 1..3: encoder level features back to NCHW ----
  float* dout = (float*)d_out;
  size_t off0 = (size_t)NB * CDIM * 16384;
  size_t off1 = off0 + (size_t)NB * CDIM * 256;
  size_t off2 = off1 + (size_t)NB * CDIM * 1024;
  const size_t outoff[3] = {off0, off1, off2};
  for (int i = 0; i < 3; ++i) {
    int hw = lh[i] * lh[i];
    outwrite(q + (size_t)lst[i] * CDIM, (size_t)STOT, dout + outoff[i], hw);
  }

  // ---- FPN: lateral conv + GN, upsample-add, 3x3 conv (9 WMMA taps) + GN + relu
  feat2a(feat[0], featA, 96, 16384);
  gemm(featA, 96, pk_lat, nullptr, nullptr, lat, NB * 16384, CDIM, 96, 0, 0, 0, 0, 0);
  k_groupnorm<<<NB * 32, 256, 0, stream>>>(lat, (size_t)16384 * CDIM, 16384, latgg, latgb, 0);
  k_upadd<<<nblk((size_t)NB * 16384 * CDIM, 256), 256, 0, stream>>>(q, lat);
  k_tobf<<<nblk((size_t)NB * 16384 * CDIM, 256), 256, 0, stream>>>(lat, yinbf, (size_t)NB * 16384 * CDIM);
  for (int t = 0; t < 9; ++t) {
    int dy = t / 3 - 1, dx = t % 3 - 1;
    gemm(yinbf, 256, pk_tap[t], nullptr, nullptr, lat, NB * 16384, CDIM, CDIM,
         GF_CONV | (t == 0 ? 0 : GF_ACCUM), 128, 128, dy, dx);
  }
  k_groupnorm<<<NB * 32, 256, 0, stream>>>(lat, (size_t)16384 * CDIM, 16384, outgg, outgb, 1);
  k_tobf<<<nblk((size_t)NB * 16384 * CDIM, 256), 256, 0, stream>>>(lat, convbf, (size_t)NB * 16384 * CDIM);
  gemm(convbf, 256, pk_mask, maskb, nullptr, maskout, NB * 16384, CDIM, CDIM, 0, 0, 0, 0, 0);
  outwrite(maskout, (size_t)16384, dout, 16384);
}
